// PolicyNetwork_67731634258681
// MI455X (gfx1250) — matrix operand order repaired and checked by reading, not compiled, not run on hardware
//
#include <hip/hip_runtime.h>

#define H   256
#define H2  512
#define NHEADS 4
#define HDIM 64

typedef _Float16 half_t;
typedef _Float16 v16h __attribute__((ext_vector_type(16)));
typedef _Float16 v8h  __attribute__((ext_vector_type(8)));
typedef float    v8f  __attribute__((ext_vector_type(8)));

// ---------------------------------------------------------------------------
// One 16x16 output tile, K-loop with V_WMMA_F32_16X16X32_F16.
// Xs: [16][K] f16 in LDS (row-major, K = 256 or 512, 32B aligned rows)
// Wg: [256][K] f16 in global (row n = output column n of x @ W^T)
// Fragment layouts (A/B: this code's contiguous k order, same for both; C/D: CDNA5 ISA 7.12.2):
//   A (16x32 f16): lane L holds M=L&15, halves i   -> K=(L>>4)*16+i
//   B (32x16 f16): lane L holds N=L&15, halves i   -> K=(L>>4)*16+i
//   C/D (16x16 f32): VGPR r, lane L -> M=r+8*(L>>4), N=L&15
// ---------------------------------------------------------------------------
__device__ __forceinline__ v8f gemm16x16(const half_t* __restrict__ Xs,
                                         const half_t* __restrict__ Wg,
                                         int K, int nt, int lane, v8f acc)
{
  const int m  = lane & 15;
  const int kb = (lane >> 4) << 3;   // 0 or 8
  const int kk = (lane >> 4) << 4;   // 0 or 16
  const half_t* xr = Xs + m * K;
  const half_t* wr = Wg + (nt * 16 + (lane & 15)) * K;
  for (int k0 = 0; k0 < K; k0 += 32) {
    v16h a = *(const v16h*)(xr + k0 + kk);
    v16h b = *(const v16h*)(wr + k0 + kk);
    acc = __builtin_amdgcn_wmma_f32_16x16x32_f16(false, a, false, b,
                                                 (short)0, acc, false, false);
  }
  return acc;
}

// --------------------------- small utility kernels -------------------------

__global__ void conv_f32_f16_kernel(const float* __restrict__ src,
                                    half_t* __restrict__ dst, int n)
{
  int i = blockIdx.x * 256 + threadIdx.x;
  if (i < n) dst[i] = (half_t)src[i];
}

__global__ void colsum_kernel(const float* __restrict__ all_emb,
                              float* __restrict__ gpart, int NA)
{
  int t = threadIdx.x;
  int rpb = (NA + gridDim.x - 1) / (int)gridDim.x;
  int r0 = blockIdx.x * rpb;
  int r1 = r0 + rpb; if (r1 > NA) r1 = NA;
  float s = 0.f;
  for (int r = r0; r < r1; ++r) s += all_emb[(size_t)r * H + t];
  gpart[blockIdx.x * H + t] = s;
}

__global__ void gmean_kernel(const float* __restrict__ gpart,
                             float* __restrict__ g, int nb, int NA)
{
  int t = threadIdx.x;
  float s = 0.f;
  for (int b = 0; b < nb; ++b) s += gpart[b * H + t];
  g[t] = s / (float)NA;
}

// query = relu(g @ qgW1^T + b1) @ qgW2^T + b2 ;  qvec = query @ Wq^T + bq
__global__ void query_kernel(const float* __restrict__ g,
                             const float* __restrict__ qgW1, const float* __restrict__ qgb1,
                             const float* __restrict__ qgW2, const float* __restrict__ qgb2,
                             const float* __restrict__ Wq,   const float* __restrict__ bq,
                             float* __restrict__ qvec)
{
  __shared__ float gl[H], h1[H], qy[H];
  int t = threadIdx.x;
  gl[t] = g[t];
  __syncthreads();
  float s = qgb1[t];
  for (int d = 0; d < H; ++d) s += gl[d] * qgW1[t * H + d];
  h1[t] = s > 0.f ? s : 0.f;
  __syncthreads();
  s = qgb2[t];
  for (int d = 0; d < H; ++d) s += h1[d] * qgW2[t * H + d];
  qy[t] = s;
  __syncthreads();
  s = bq[t];
  for (int d = 0; d < H; ++d) s += qy[d] * Wq[t * H + d];
  qvec[t] = s;
}

// ------------------------------ fused WMMA core ----------------------------

__global__ void __launch_bounds__(256)
keys_kernel(const float* __restrict__ embA, const float* __restrict__ embB,
            const int* __restrict__ idxA, const int* __restrict__ idxB,
            const half_t* __restrict__ W1h, const float* __restrict__ b1,
            const half_t* __restrict__ W2h, const float* __restrict__ b2,
            const half_t* __restrict__ Wkh, const float* __restrict__ bk,
            const half_t* __restrict__ Wvh, const float* __restrict__ bv,
            const float* __restrict__ qvec,
            half_t* __restrict__ keys_out, half_t* __restrict__ v_out,
            float* __restrict__ scores, int rowOffset, int Atot)
{
  __shared__ __align__(32) half_t Xs[16 * H2];  // gathered input, f16
  __shared__ __align__(32) half_t Hs[16 * H];   // hidden after relu, f16
  __shared__ __align__(32) half_t Ks[16 * H];   // keys tile, f16
  __shared__ float KT[16 * H];                  // k-projection, f32
  __shared__ float qs[H];
  __shared__ int   ia[16], ib[16];

  const int t    = threadIdx.x;
  const int rb   = blockIdx.x * 16;       // local row base within this variant
  const int grow = rowOffset + rb;        // global action row base

  if (t < 16) { ia[t] = idxA[rb + t]; ib[t] = idxB[rb + t]; }
  qs[t] = qvec[t];
  __syncthreads();

  // Gather + f32->f16: x = concat(embA[idxA], embB[idxB])  (16 x 512)
#pragma unroll
  for (int j = 0; j < 32; ++j) {
    int e   = t + j * 256;
    int row = e >> 9, col = e & 511;
    float v = (col < H) ? embA[ia[row] * H + col]
                        : embB[ib[row] * H + (col - H)];
    Xs[row * H2 + col] = (half_t)v;
  }
  __syncthreads();

  const int wave = t >> 5, lane = t & 31;
  const int n  = lane & 15;
  const int mh = (lane >> 4) << 3;        // 0 or 8

  // Phase 1: H1 = relu(X @ W1^T + b1)
#pragma unroll
  for (int s = 0; s < 2; ++s) {
    int nt = wave * 2 + s;
    v8f acc = {};
    acc = gemm16x16(Xs, W1h, H2, nt, lane, acc);
    float bias = b1[nt * 16 + n];
#pragma unroll
    for (int r = 0; r < 8; ++r) {
      float v = acc[r] + bias;
      Hs[(mh + r) * H + nt * 16 + n] = (half_t)(v > 0.f ? v : 0.f);
    }
  }
  __syncthreads();

  // Phase 2: keys = H1 @ W2^T + b2  (-> LDS f16 + global f16)
#pragma unroll
  for (int s = 0; s < 2; ++s) {
    int nt = wave * 2 + s;
    v8f acc = {};
    acc = gemm16x16(Hs, W2h, H, nt, lane, acc);
    float bias = b2[nt * 16 + n];
#pragma unroll
    for (int r = 0; r < 8; ++r) {
      half_t hv = (half_t)(acc[r] + bias);
      Ks[(mh + r) * H + nt * 16 + n] = hv;
      keys_out[(size_t)(grow + mh + r) * H + nt * 16 + n] = hv;
    }
  }
  __syncthreads();

  // Phase 3: k = keys @ Wk^T + bk  (-> LDS f32 for scores)
#pragma unroll
  for (int s = 0; s < 2; ++s) {
    int nt = wave * 2 + s;
    v8f acc = {};
    acc = gemm16x16(Ks, Wkh, H, nt, lane, acc);
    float bias = bk[nt * 16 + n];
#pragma unroll
    for (int r = 0; r < 8; ++r)
      KT[(mh + r) * H + nt * 16 + n] = acc[r] + bias;
  }

  // Phase 4: v = keys @ Wv^T + bv  (-> global f16)
#pragma unroll
  for (int s = 0; s < 2; ++s) {
    int nt = wave * 2 + s;
    v8f acc = {};
    acc = gemm16x16(Ks, Wvh, H, nt, lane, acc);
    float bias = bv[nt * 16 + n];
#pragma unroll
    for (int r = 0; r < 8; ++r)
      v_out[(size_t)(grow + mh + r) * H + nt * 16 + n] = (half_t)(acc[r] + bias);
  }
  __syncthreads();

  // scores[h][grow+row] = (q_h . k_h(row)) / sqrt(64)
  if (t < 64) {
    int row = t & 15, h = t >> 4;
    float s = 0.f;
    for (int d = 0; d < HDIM; ++d)
      s += qs[h * HDIM + d] * KT[row * H + h * HDIM + d];
    scores[(size_t)h * Atot + grow + row] = s * 0.125f;
  }
}

// --------------------------- softmax reductions ----------------------------

// Online (max, sum-exp) per block chunk. x: [rows][n] row-major;
// blockIdx.y = row, blockIdx.x = chunk. pairs: [rows][nbpr][2].
__global__ void softmax_stats_kernel(const float* __restrict__ x,
                                     float* __restrict__ pairs,
                                     int n, int nbpr)
{
  __shared__ float sm[256], ss[256];
  int t = threadIdx.x;
  const float* xr = x + (size_t)blockIdx.y * n;
  int chunk = (n + nbpr - 1) / nbpr;
  int a0 = blockIdx.x * chunk;
  int a1 = a0 + chunk; if (a1 > n) a1 = n;
  float m = -3.0e38f, s = 0.f;
  for (int i = a0 + t; i < a1; i += 256) {
    float v = xr[i];
    if (v > m) { s = s * expf(m - v) + 1.f; m = v; }
    else         s += expf(v - m);
  }
  sm[t] = m; ss[t] = s;
  __syncthreads();
  for (int off = 128; off > 0; off >>= 1) {
    if (t < off) {
      float m2 = sm[t + off], s2 = ss[t + off];
      float mm = sm[t],      s1 = ss[t];
      if (m2 > mm) { s1 = s1 * expf(mm - m2) + s2; mm = m2; }
      else           s1 += s2 * expf(m2 - mm);
      sm[t] = mm; ss[t] = s1;
    }
    __syncthreads();
  }
  if (t == 0) {
    pairs[((size_t)blockIdx.y * nbpr + blockIdx.x) * 2 + 0] = sm[0];
    pairs[((size_t)blockIdx.y * nbpr + blockIdx.x) * 2 + 1] = ss[0];
  }
}

__global__ void stats_combine_kernel(const float* __restrict__ pairs,
                                     float* __restrict__ out, int rows, int nb)
{
  int r = threadIdx.x;
  if (r < rows) {
    float m = -3.0e38f, s = 0.f;
    for (int i = 0; i < nb; ++i) {
      float m2 = pairs[((size_t)r * nb + i) * 2 + 0];
      float s2 = pairs[((size_t)r * nb + i) * 2 + 1];
      if (m2 > m) { s = s * expf(m - m2) + s2; m = m2; }
      else          s += s2 * expf(m2 - m);
    }
    out[r * 2 + 0] = m;
    out[r * 2 + 1] = s;
  }
}

// ------------------------------- attention ---------------------------------

// Partial ctx[h][d] over a row chunk; thread t owns (h=t/64, d=t%64) == col t.
__global__ void ctx_partial_kernel(const float* __restrict__ scores,
                                   const half_t* __restrict__ vh,
                                   const float* __restrict__ headMS,
                                   float* __restrict__ ctxp, int Atot)
{
  int t = threadIdx.x;
  int h = t >> 6;
  float m = headMS[h * 2 + 0];
  float invS = 1.f / headMS[h * 2 + 1];
  int chunk = (Atot + (int)gridDim.x - 1) / (int)gridDim.x;
  int a0 = blockIdx.x * chunk;
  int a1 = a0 + chunk; if (a1 > Atot) a1 = Atot;
  float acc = 0.f;
  for (int a = a0; a < a1; ++a) {
    float w = expf(scores[(size_t)h * Atot + a] - m) * invS;
    acc += w * (float)vh[(size_t)a * H + t];
  }
  ctxp[blockIdx.x * H + t] = acc;
}

__global__ void ctx_finish_kernel(const float* __restrict__ ctxp, int nb,
                                  const float* __restrict__ Wo,
                                  const float* __restrict__ bo,
                                  float* __restrict__ attn_out)
{
  __shared__ float ctx[H];
  int t = threadIdx.x;
  float s = 0.f;
  for (int b = 0; b < nb; ++b) s += ctxp[b * H + t];
  ctx[t] = s;
  __syncthreads();
  float o = bo[t];
  for (int d = 0; d < H; ++d) o += Wo[t * H + d] * ctx[d];
  attn_out[t] = o;
}

// logits[a] = keys[a] . attn_out ; 32 rows per block, 8 threads per row
__global__ void logits_kernel(const half_t* __restrict__ keys_h,
                              const float* __restrict__ attn_out,
                              float* __restrict__ logits, int Atot)
{
  __shared__ float ao[H];
  __shared__ float part[256];
  int t = threadIdx.x;
  ao[t] = attn_out[t];
  __syncthreads();
  int r = t >> 3, c8 = t & 7;
  int row = blockIdx.x * 32 + r;
  float s = 0.f;
  if (row < Atot) {
    const half_t* kr = keys_h + (size_t)row * H + c8 * 32;
#pragma unroll
    for (int j = 0; j < 32; ++j) s += (float)kr[j] * ao[c8 * 32 + j];
  }
  part[t] = s;
  __syncthreads();
  if (t < 32) {
    int row2 = blockIdx.x * 32 + t;
    if (row2 < Atot) {
      float tot = 0.f;
#pragma unroll
      for (int j = 0; j < 8; ++j) tot += part[t * 8 + j];
      logits[row2] = tot;
    }
  }
}

__global__ void probs_kernel(const float* __restrict__ logits,
                             const float* __restrict__ MS,
                             float* __restrict__ probs, int Atot)
{
  int a = blockIdx.x * 256 + threadIdx.x;
  if (a < Atot) probs[a] = expf(logits[a] - MS[0]) / MS[1];
}

// ------------------------------- launcher ----------------------------------

extern "C" void kernel_launch(void* const* d_in, const int* in_sizes, int n_in,
                              void* d_out, int out_size, void* d_ws, size_t ws_size,
                              hipStream_t stream)
{
  const float* qubit_emb = (const float*)d_in[0];
  const float* qpu_emb   = (const float*)d_in[1];
  const float* gate_emb  = (const float*)d_in[2];
  const float* all_emb   = (const float*)d_in[3];
  const float* time_tab  = (const float*)d_in[4];
  const float* map_W1 = (const float*)d_in[5];  const float* map_b1 = (const float*)d_in[6];
  const float* map_W2 = (const float*)d_in[7];  const float* map_b2 = (const float*)d_in[8];
  const float* sch_W1 = (const float*)d_in[9];  const float* sch_b1 = (const float*)d_in[10];
  const float* sch_W2 = (const float*)d_in[11]; const float* sch_b2 = (const float*)d_in[12];
  const float* qg_W1  = (const float*)d_in[13]; const float* qg_b1  = (const float*)d_in[14];
  const float* qg_W2  = (const float*)d_in[15]; const float* qg_b2  = (const float*)d_in[16];
  const float* Wq = (const float*)d_in[17]; const float* bq = (const float*)d_in[18];
  const float* Wk = (const float*)d_in[19]; const float* bk = (const float*)d_in[20];
  const float* Wv = (const float*)d_in[21]; const float* bv = (const float*)d_in[22];
  const float* Wo = (const float*)d_in[23]; const float* bo = (const float*)d_in[24];
  const int* map_qubit  = (const int*)d_in[25];
  const int* map_qpu    = (const int*)d_in[26];
  const int* sched_gate = (const int*)d_in[27];
  const int* sched_time = (const int*)d_in[28];

  const int Nm   = in_sizes[25];
  const int Ns   = in_sizes[27];
  const int Atot = Nm + Ns;
  const int NA   = in_sizes[3] / H;
  (void)n_in; (void)out_size; (void)ws_size;

  char* w = (char*)d_ws;
  size_t off = 0;
  auto take = [&](size_t bytes) -> void* {
    void* p = w + off;
    off += (bytes + 255) & ~(size_t)255;
    return p;
  };
  half_t* keys_h = (half_t*)take((size_t)Atot * H * sizeof(half_t));
  half_t* v_h    = (half_t*)take((size_t)Atot * H * sizeof(half_t));
  float*  scores = (float*)take((size_t)NHEADS * Atot * sizeof(float));
  half_t* W1m_h  = (half_t*)take((size_t)H * H2 * sizeof(half_t));
  half_t* W2m_h  = (half_t*)take((size_t)H * H  * sizeof(half_t));
  half_t* W1s_h  = (half_t*)take((size_t)H * H2 * sizeof(half_t));
  half_t* W2s_h  = (half_t*)take((size_t)H * H  * sizeof(half_t));
  half_t* Wk_h   = (half_t*)take((size_t)H * H  * sizeof(half_t));
  half_t* Wv_h   = (half_t*)take((size_t)H * H  * sizeof(half_t));
  float*  gpart  = (float*)take(256 * H * sizeof(float));
  float*  g      = (float*)take(H * sizeof(float));
  float*  qvec   = (float*)take(H * sizeof(float));
  float*  hpairs = (float*)take(NHEADS * 64 * 2 * sizeof(float));
  float*  headMS = (float*)take(NHEADS * 2 * sizeof(float));
  float*  ctxp   = (float*)take(256 * H * sizeof(float));
  float*  aout   = (float*)take(H * sizeof(float));
  float*  lpairs = (float*)take(64 * 2 * sizeof(float));
  float*  lMS    = (float*)take(2 * sizeof(float));

  float* out_probs  = (float*)d_out;
  float* out_logits = out_probs + Atot;

  // 1. Weight conversions to f16 (tiny)
  conv_f32_f16_kernel<<<(H * H2 + 255) / 256, 256, 0, stream>>>(map_W1, W1m_h, H * H2);
  conv_f32_f16_kernel<<<(H * H  + 255) / 256, 256, 0, stream>>>(map_W2, W2m_h, H * H);
  conv_f32_f16_kernel<<<(H * H2 + 255) / 256, 256, 0, stream>>>(sch_W1, W1s_h, H * H2);
  conv_f32_f16_kernel<<<(H * H  + 255) / 256, 256, 0, stream>>>(sch_W2, W2s_h, H * H);
  conv_f32_f16_kernel<<<(H * H  + 255) / 256, 256, 0, stream>>>(Wk, Wk_h, H * H);
  conv_f32_f16_kernel<<<(H * H  + 255) / 256, 256, 0, stream>>>(Wv, Wv_h, H * H);

  // 2. g = mean(all_embeddings), deterministic 2-stage
  colsum_kernel<<<256, 256, 0, stream>>>(all_emb, gpart, NA);
  gmean_kernel<<<1, 256, 0, stream>>>(gpart, g, 256, NA);

  // 3. query MLP + q projection
  query_kernel<<<1, 256, 0, stream>>>(g, qg_W1, qg_b1, qg_W2, qg_b2, Wq, bq, qvec);

  // 4. Fused WMMA: keys / k-scores / v for both action families
  keys_kernel<<<Nm / 16, 256, 0, stream>>>(
      qubit_emb, qpu_emb, map_qubit, map_qpu,
      W1m_h, map_b1, W2m_h, map_b2, Wk_h, bk, Wv_h, bv, qvec,
      keys_h, v_h, scores, 0, Atot);
  keys_kernel<<<Ns / 16, 256, 0, stream>>>(
      gate_emb, time_tab, sched_gate, sched_time,
      W1s_h, sch_b1, W2s_h, sch_b2, Wk_h, bk, Wv_h, bv, qvec,
      keys_h, v_h, scores, Nm, Atot);

  // 5. Attention softmax over A per head (deterministic 2-stage online)
  dim3 hs(64, NHEADS);
  softmax_stats_kernel<<<hs, 256, 0, stream>>>(scores, hpairs, Atot, 64);
  stats_combine_kernel<<<1, 32, 0, stream>>>(hpairs, headMS, NHEADS, 64);

  // 6. ctx = sum_a w[h,a] * v[a,h,:] ; attn_out = Wo ctx + bo
  ctx_partial_kernel<<<256, 256, 0, stream>>>(scores, v_h, headMS, ctxp, Atot);
  ctx_finish_kernel<<<1, 256, 0, stream>>>(ctxp, 256, Wo, bo, aout);

  // 7. logits, softmax -> probs
  logits_kernel<<<(Atot + 31) / 32, 256, 0, stream>>>(keys_h, aout, out_logits, Atot);
  dim3 ls(64, 1);
  softmax_stats_kernel<<<ls, 256, 0, stream>>>(out_logits, lpairs, Atot, 64);
  stats_combine_kernel<<<1, 32, 0, stream>>>(lpairs, lMS, 1, 64);
  probs_kernel<<<(Atot + 255) / 256, 256, 0, stream>>>(out_logits, lMS, out_probs, Atot);
}
